// TemporalGNN_82394652606710
// MI455X (gfx1250) — matrix operand order repaired and checked by reading, not compiled, not run on hardware
//
#include <hip/hip_runtime.h>

#define N_NODES  50000
#define N_EDGES  1600000
#define NODE_DIM 128
#define EDGE_DIM 16
#define HIDDEN   128

typedef __attribute__((ext_vector_type(16))) _Float16 v16h;
typedef __attribute__((ext_vector_type(8)))  float    v8f;

// ---------------------------------------------------------------- zero
__global__ void k_zero(float* __restrict__ p, long n) {
  long i = (long)blockIdx.x * blockDim.x + threadIdx.x;
  if (i < n) p[i] = 0.f;
}

// ------------------------------------------- layer-invariant edge stats
// Eagg[v][k] = sum over edges with dst==v of edge_attr[e][k]; deg[v] = #edges
__global__ void k_edge_agg(const long long* __restrict__ ei,
                           const float* __restrict__ ea,
                           float* __restrict__ Eagg,
                           float* __restrict__ deg) {
  long t = (long)blockIdx.x * blockDim.x + threadIdx.x;
  if (t >= (long)N_EDGES * EDGE_DIM) return;
  long e = t >> 4;
  int  k = (int)(t & 15);
  int  dst = (int)ei[(long)N_EDGES + e];
  atomicAdd(&Eagg[(long)dst * EDGE_DIM + k], ea[t]);
  if (k == 0) atomicAdd(&deg[dst], 1.0f);
}

// ------------------------------------------------- per-layer h scatter
// A[dst] += h[src]; 32 lanes per edge, float4 per lane (coalesced 512B row)
__global__ void k_scatter_h(const long long* __restrict__ ei,
                            const float* __restrict__ h,
                            float* __restrict__ A) {
  long t = (long)blockIdx.x * blockDim.x + threadIdx.x;
  if (t >= (long)N_EDGES * 32) return;
  long e = t >> 5;
  int  c = (int)(t & 31);
  int  src = (int)ei[e];
  int  dst = (int)ei[(long)N_EDGES + e];
  const float4 v = ((const float4*)(h + (long)src * HIDDEN))[c];
  float* ap = A + (long)dst * HIDDEN + c * 4;
  atomicAdd(ap + 0, v.x);
  atomicAdd(ap + 1, v.y);
  atomicAdd(ap + 2, v.z);
  atomicAdd(ap + 3, v.w);
}

// --------------------------------------------------------- weight pack
// Pack [W1 (k1 x 128); W2 (k2 x 128); zero-pad] into WMMA B-fragment
// layout: frag index (kt*8+nt)*32+lane, 16 halves per lane,
// half slot j of lane -> K = kt*32 + (lane/16)*16 + j, col = nt*16 + lane%16.
__global__ void k_pack_w(const float* __restrict__ W1, int k1,
                         const float* __restrict__ W2, int k2,
                         _Float16* __restrict__ out, int nkt) {
  int t = blockIdx.x * blockDim.x + threadIdx.x;
  if (t >= nkt * 256) return;
  int kt   = t >> 8;
  int rem  = t & 255;
  int nt   = rem >> 5;
  int lane = rem & 31;
  int halfk = (lane >> 4) * 16;
  int col   = lane & 15;
  v16h frag;
#pragma unroll
  for (int j = 0; j < 16; ++j) {
    int K = kt * 32 + halfk + j;
    int n = nt * 16 + col;
    float v = 0.f;
    if (K < k1)            v = W1[(long)K * HIDDEN + n];
    else if (K < k1 + k2)  v = W2[(long)(K - k1) * HIDDEN + n];
    frag[j] = (_Float16)v;
  }
  ((v16h*)out)[(kt * 8 + nt) * 32 + lane] = frag;
}

__global__ void k_bias(const float* __restrict__ b1,
                       const float* __restrict__ b2,
                       float* __restrict__ out) {
  int t = threadIdx.x;
  if (t < HIDDEN) out[t] = b1[t] + (b2 ? b2[t] : 0.f);
}

// ------------------------------------------------------ WMMA GEMM core
// out[r][c] = relu( sum_k Afeat[r][k]*W[k][c] (+ Eagg part) + dscale(r)*bias[c] )
// One wave per 16-row tile, 8 column tiles of 16, K = 4x32 (+1x32 edge tile).
__global__ __launch_bounds__(256) void k_gemm_wmma(
    const float* __restrict__ Afeat,     // n_rows x 128 (f32)
    const float* __restrict__ Eagg,      // n_rows x 16 (f32) or nullptr
    const _Float16* __restrict__ Wp,     // packed B fragments
    const float* __restrict__ deg,       // n_rows or nullptr (-> 1.0)
    const float* __restrict__ bias,      // 128
    float* __restrict__ out,             // n_rows x 128
    int n_rows) {
  const v16h* Wv = (const v16h*)Wp;
  int wave = (int)(((long)blockIdx.x * blockDim.x + threadIdx.x) >> 5);
  int lane = threadIdx.x & 31;
  int row0 = wave * 16;
  if (row0 >= n_rows) return;           // whole wave exits; EXEC stays all-1s for WMMA
  int halfo = (lane >> 4) * 8;          // K-subgroup for A frags / row offset for D
  int l16   = lane & 15;
  const float* aBase = Afeat + (long)(row0 + l16) * HIDDEN;

  v8f acc[8] = {};

#pragma unroll
  for (int kt = 0; kt < 4; ++kt) {
    v16h af;
    const float* p0 = aBase + kt * 32 + halfo * 2;   // K = kt*32 + half*16 + (0..7)
    const float* p1 = p0 + 8;                    // K = kt*32 + half*16 + 8 + (0..7)
#pragma unroll
    for (int j = 0; j < 8; ++j) af[j]     = (_Float16)p0[j];
#pragma unroll
    for (int j = 0; j < 8; ++j) af[8 + j] = (_Float16)p1[j];
#pragma unroll
    for (int nt = 0; nt < 8; ++nt) {
      v16h bf = Wv[(kt * 8 + nt) * 32 + lane];
      acc[nt] = __builtin_amdgcn_wmma_f32_16x16x32_f16(
          false, af, false, bf, (short)0, acc[nt], false, false);
    }
  }

  if (Eagg != nullptr) {                // K tile 4: [Eagg (16) | zero-pad (16)]
    v16h af;
    const float* p0 = Eagg + (long)(row0 + l16) * EDGE_DIM;
#pragma unroll
    for (int j = 0; j < 8; ++j) af[j]     = (_Float16)(halfo ? 0.f : p0[j]);
#pragma unroll
    for (int j = 0; j < 8; ++j) af[8 + j] = (_Float16)(halfo ? 0.f : p0[8 + j]);
#pragma unroll
    for (int nt = 0; nt < 8; ++nt) {
      v16h bf = Wv[(4 * 8 + nt) * 32 + lane];
      acc[nt] = __builtin_amdgcn_wmma_f32_16x16x32_f16(
          false, af, false, bf, (short)0, acc[nt], false, false);
    }
  }

  // Epilogue: D tile layout = VGPR i: row = halfo + i, col = nt*16 + l16
  float dval[8];
#pragma unroll
  for (int i = 0; i < 8; ++i) {
    int r = row0 + halfo + i;
    dval[i] = deg ? deg[r] : 1.0f;
  }
#pragma unroll
  for (int nt = 0; nt < 8; ++nt) {
    float bcol = bias[nt * 16 + l16];
#pragma unroll
    for (int i = 0; i < 8; ++i) {
      float v = acc[nt][i] + dval[i] * bcol;
      v = v > 0.f ? v : 0.f;
      out[(long)(row0 + halfo + i) * HIDDEN + nt * 16 + l16] = v;
    }
  }
}

// ---------------------------------------------------------- final head
__global__ void k_head(const float* __restrict__ hid,
                       const float* __restrict__ Wc2,
                       const float* __restrict__ bc2,
                       float* __restrict__ out) {
  __shared__ float s[HIDDEN];
  int v = blockIdx.x;
  int t = threadIdx.x;
  s[t] = hid[(long)v * HIDDEN + t] * Wc2[t];
  __syncthreads();
  for (int off = 64; off > 0; off >>= 1) {
    if (t < off) s[t] += s[t + off];
    __syncthreads();
  }
  if (t == 0) out[v] = s[0] + bc2[0];
}

// ---------------------------------------------------------------- host
extern "C" void kernel_launch(void* const* d_in, const int* in_sizes, int n_in,
                              void* d_out, int out_size, void* d_ws, size_t ws_size,
                              hipStream_t stream) {
  (void)in_sizes; (void)n_in; (void)out_size; (void)ws_size;
  const float*     x   = (const float*)d_in[0];
  const long long* ei  = (const long long*)d_in[1];
  const float*     ea  = (const float*)d_in[2];
  const float* Wn[3] = {(const float*)d_in[3], (const float*)d_in[7],  (const float*)d_in[11]};
  const float* bn[3] = {(const float*)d_in[4], (const float*)d_in[8],  (const float*)d_in[12]};
  const float* We[3] = {(const float*)d_in[5], (const float*)d_in[9],  (const float*)d_in[13]};
  const float* be[3] = {(const float*)d_in[6], (const float*)d_in[10], (const float*)d_in[14]};
  const float* Wc1 = (const float*)d_in[15];
  const float* bc1 = (const float*)d_in[16];
  const float* Wc2 = (const float*)d_in[17];
  const float* bc2 = (const float*)d_in[18];
  float* out = (float*)d_out;

  char* wsp = (char*)d_ws;
  auto take = [&](size_t bytes) -> char* {
    char* p = wsp; wsp += (bytes + 255) & ~(size_t)255; return p;
  };
  float*    Eagg    = (float*)take((size_t)N_NODES * EDGE_DIM * 4);
  float*    deg     = (float*)take((size_t)N_NODES * 4);
  float*    A       = (float*)take((size_t)N_NODES * HIDDEN * 4);
  float*    h0      = (float*)take((size_t)N_NODES * HIDDEN * 4);
  float*    h1      = (float*)take((size_t)N_NODES * HIDDEN * 4);
  _Float16* Wpack   = (_Float16*)take((size_t)5 * 8 * 32 * 16 * 2);
  float*    biasBuf = (float*)take(HIDDEN * 4);

  auto cdiv = [](long a, long b) { return (unsigned)((a + b - 1) / b); };

  // Layer-invariant edge aggregation (re-zeroed every call: deterministic work)
  k_zero<<<cdiv((long)N_NODES * EDGE_DIM, 256), 256, 0, stream>>>(Eagg, (long)N_NODES * EDGE_DIM);
  k_zero<<<cdiv((long)N_NODES, 256), 256, 0, stream>>>(deg, (long)N_NODES);
  k_edge_agg<<<cdiv((long)N_EDGES * EDGE_DIM, 256), 256, 0, stream>>>(ei, ea, Eagg, deg);

  const unsigned gemm_blocks = cdiv((long)N_NODES / 16, 8);  // 8 waves/block
  const float* h_in = x;
  float* bufs[2] = {h0, h1};

  for (int L = 0; L < 3; ++L) {
    float* h_out = bufs[L & 1];
    k_zero<<<cdiv((long)N_NODES * HIDDEN, 256), 256, 0, stream>>>(A, (long)N_NODES * HIDDEN);
    k_scatter_h<<<cdiv((long)N_EDGES * 32, 256), 256, 0, stream>>>(ei, h_in, A);
    k_pack_w<<<5, 256, 0, stream>>>(Wn[L], NODE_DIM, We[L], EDGE_DIM, Wpack, 5);
    k_bias<<<1, 128, 0, stream>>>(bn[L], be[L], biasBuf);
    k_gemm_wmma<<<gemm_blocks, 256, 0, stream>>>(A, Eagg, Wpack, deg, biasBuf, h_out, N_NODES);
    h_in = h_out;
  }

  // head: hid = relu(h @ Wc1 + bc1); logits = hid @ Wc2 + bc2
  float* hid = bufs[1];  // free buffer after layer 3 (which wrote bufs[0])
  k_pack_w<<<4, 256, 0, stream>>>(Wc1, HIDDEN, nullptr, 0, Wpack, 4);
  k_bias<<<1, 128, 0, stream>>>(bc1, nullptr, biasBuf);
  k_gemm_wmma<<<gemm_blocks, 256, 0, stream>>>(h_in, nullptr, Wpack, nullptr, biasBuf, hid, N_NODES);
  k_head<<<N_NODES, 128, 0, stream>>>(hid, Wc2, bc2, out);
}
